// QLinearLayer_50233937494448
// MI455X (gfx1250) — 8-bit matrix operand order repaired and checked by reading, compiled, not run on hardware
//
#include <hip/hip_runtime.h>
#include <hip/hip_bf16.h>


typedef __attribute__((ext_vector_type(8))) int v8i;
typedef int vint4 __attribute__((vector_size(16)));
typedef __attribute__((address_space(1))) vint4 g_vint4;
typedef __attribute__((address_space(3))) vint4 l_vint4;

#define MDIM 1024
#define NDIM 11008
#define KDIM 4096

#if defined(__has_builtin)
#if __has_builtin(__builtin_amdgcn_global_load_async_to_lds_b128)
#define USE_ASYNC_LDS 1
#endif
#endif
#ifndef USE_ASYNC_LDS
#define USE_ASYNC_LDS 0
#endif

#if defined(__has_builtin)
#if __has_builtin(__builtin_amdgcn_s_wait_asynccnt)
#define WAIT_ASYNCCNT(n) __builtin_amdgcn_s_wait_asynccnt(n)
#endif
#endif
#ifndef WAIT_ASYNCCNT
#define WAIT_ASYNCCNT(n) asm volatile("s_wait_asynccnt %0" ::"i"(n) : "memory")
#endif

__global__ __launch_bounds__(256) void pack_i8_kernel(
    const int* __restrict__ in, unsigned int* __restrict__ out, int n4) {
  int i = blockIdx.x * blockDim.x + threadIdx.x;
  if (i < n4) {
    const int4 v = reinterpret_cast<const int4*>(in)[i];
    unsigned int p = (unsigned int)(v.x & 0xff)
                   | ((unsigned int)(v.y & 0xff) << 8)
                   | ((unsigned int)(v.z & 0xff) << 16)
                   | ((unsigned int)(v.w & 0xff) << 24);
    out[i] = p;
  }
}

__device__ __forceinline__ void compute_stage(
    const int* __restrict__ Asb, const int* __restrict__ Bsb,
    int wm, int wn, int ln, int half, v8i acc[2][4]) {
  #pragma unroll
  for (int ks = 0; ks < 2; ++ks) {
    const int ksd = ks * 16;
    v8i a[2], b[4];

    #pragma unroll
    for (int mi = 0; mi < 2; ++mi) {
      const int rbase = (wm * 32 + mi * 16 + ln) * 32 + ksd + half * 2;
      #pragma unroll
      for (int j = 0; j < 8; ++j)
        a[mi][j] = Asb[rbase + (j >> 1) * 4 + (j & 1)];
    }

    #pragma unroll
    for (int ni = 0; ni < 4; ++ni) {
      const int cbase = (wn * 64 + ni * 16 + ln) * 32 + ksd + half * 2;
      #pragma unroll
      for (int j = 0; j < 8; ++j)
        b[ni][j] = Bsb[cbase + (j >> 1) * 4 + (j & 1)];
    }

    #pragma unroll
    for (int mi = 0; mi < 2; ++mi)
      #pragma unroll
      for (int ni = 0; ni < 4; ++ni)
        acc[mi][ni] = __builtin_amdgcn_wmma_i32_16x16x64_iu8(
             false, a[mi],  false, b[ni],
            acc[mi][ni],  false,  false);
  }
}

__global__ __launch_bounds__(256) void qgemm_i8_kernel(
    const unsigned char* __restrict__ qx8,
    const unsigned char* __restrict__ w8,
    const float* __restrict__ scale_x,
    const float* __restrict__ scale_w,
    const float* __restrict__ bias,
    float* __restrict__ out)
{
#if USE_ASYNC_LDS
  __shared__ int As[2][128 * 32];
  __shared__ int Bs[2][128 * 32];
#else
  __shared__ int As[1][128 * 32];
  __shared__ int Bs[1][128 * 32];
#endif

  const int tid  = threadIdx.x;
  const int lane = tid & 31;
  const int wave = tid >> 5;
  const int wm   = wave >> 1;
  const int wn   = wave & 1;
  const int half = lane >> 4;
  const int ln   = lane & 15;

  const int m0 = blockIdx.y * 128;
  const int n0 = blockIdx.x * 128;

  const int ldRow  = tid >> 3;
  const int ldColB = (tid & 7) * 16;

  v8i acc[2][4] = {};

#if USE_ASYNC_LDS
  auto issue_stage = [&](int kb, int buf) {
    #pragma unroll
    for (int r = 0; r < 4; ++r) {
      const int row = r * 32 + ldRow;
      const int di  = row * 32 + (tid & 7) * 4;
      __builtin_amdgcn_global_load_async_to_lds_b128(
          (g_vint4*)(qx8 + (size_t)(m0 + row) * KDIM + kb + ldColB),
          (l_vint4*)&As[buf][di], 0, 0);
      __builtin_amdgcn_global_load_async_to_lds_b128(
          (g_vint4*)(w8 + (size_t)(n0 + row) * KDIM + kb + ldColB),
          (l_vint4*)&Bs[buf][di], 0, 0);
    }
  };

  issue_stage(0, 0);
  for (int t = 0; t < KDIM / 128; ++t) {
    const int buf = t & 1;
    WAIT_ASYNCCNT(0);
    __syncthreads();
    if (t + 1 < KDIM / 128)
      issue_stage((t + 1) * 128, buf ^ 1);
    compute_stage(As[buf], Bs[buf], wm, wn, ln, half, acc);
  }
#else
  for (int kb = 0; kb < KDIM; kb += 128) {
    __syncthreads();
    #pragma unroll
    for (int r = 0; r < 4; ++r) {
      const int row = r * 32 + ldRow;
      const int4 va = *reinterpret_cast<const int4*>(
          qx8 + (size_t)(m0 + row) * KDIM + kb + ldColB);
      *reinterpret_cast<int4*>(&As[0][row * 32 + (tid & 7) * 4]) = va;
      const unsigned char* pb = w8 + (size_t)(n0 + row) * KDIM + kb + ldColB;
      const int4 vb = *reinterpret_cast<const int4*>(pb);
      *reinterpret_cast<int4*>(&Bs[0][row * 32 + (tid & 7) * 4]) = vb;
      __builtin_prefetch(pb + 128, 0, 1);
    }
    __syncthreads();
    compute_stage(As[0], Bs[0], wm, wn, ln, half, acc);
  }
#endif

  #pragma unroll
  for (int mi = 0; mi < 2; ++mi) {
    const int mbase = m0 + wm * 32 + mi * 16 + half * 8;
    float sx[8];
    #pragma unroll
    for (int v = 0; v < 8; ++v) sx[v] = scale_x[mbase + v];
    #pragma unroll
    for (int ni = 0; ni < 4; ++ni) {
      const int nn = n0 + wn * 64 + ni * 16 + ln;
      const float sw = scale_w[nn];
      const float bb = bias[nn];
      #pragma unroll
      for (int v = 0; v < 8; ++v)
        out[(size_t)(mbase + v) * NDIM + nn] =
            (float)acc[mi][ni][v] * sx[v] * sw + bb;
    }
  }
}

extern "C" void kernel_launch(void* const* d_in, const int* in_sizes, int n_in,
                              void* d_out, int out_size, void* d_ws, size_t ws_size,
                              hipStream_t stream) {
  const int*   qx      = (const int*)d_in[0];
  const float* scale_x = (const float*)d_in[1];
  const int*   W       = (const int*)d_in[2];
  const float* scale_w = (const float*)d_in[3];
  const float* bias    = (const float*)d_in[4];
  float*       out     = (float*)d_out;

  unsigned char* qx8 = (unsigned char*)d_ws;
  unsigned char* w8  = qx8 + (size_t)MDIM * KDIM;

  const int n4x = MDIM * KDIM / 4;
  const int n4w = NDIM * KDIM / 4;
  pack_i8_kernel<<<(n4x + 255) / 256, 256, 0, stream>>>(qx, (unsigned int*)qx8, n4x);
  pack_i8_kernel<<<(n4w + 255) / 256, 256, 0, stream>>>(W,  (unsigned int*)w8,  n4w);

  dim3 grid(NDIM / 128, MDIM / 128);
  qgemm_i8_kernel<<<grid, 256, 0, stream>>>(qx8, w8, scale_x, scale_w, bias, out);
}
